// CrossContextCluster_9405978378738
// MI455X (gfx1250) — matrix operand order repaired and checked by reading, not compiled, not run on hardware
//
#include <hip/hip_runtime.h>
#include <hip/hip_bf16.h>
#include <math.h>

typedef __attribute__((ext_vector_type(16))) _Float16 v16h;
typedef __attribute__((ext_vector_type(8)))  float    v8f;

#define WMMA_F16(a, b, c) \
  __builtin_amdgcn_wmma_f32_16x16x32_f16(false, (a), false, (b), (short)0, (c), false, false)

// Wave index forced wave-uniform (scalar) so tile bases stay in SGPRs and the
// inner-loop loads use SADDR + scalar increments instead of per-lane 64-bit
// VALU address chains (which also caused WMMA->VALU WAR v_nop bubbles).
static __device__ __forceinline__ int uwave() {
  return (int)blockIdx.x * ((int)blockDim.x >> 5) +
         __builtin_amdgcn_readfirstlane((int)threadIdx.x >> 5);
}

// ---------------------------------------------------------------------------
// Pack fp32 weight matrix W[K][N] (row-major) into f16 B-fragment layout:
// tile index t = kc*(N/16)+nt, per tile 32 lanes x 16 halfs:
//   lane c  (0..15): column nt*16+c, K = kc*32 + 0..15
//   lane c+16      : column nt*16+c, K = kc*32 + 16..31
// ---------------------------------------------------------------------------
__global__ void pack_w_bfrag(const float* __restrict__ W, _Float16* __restrict__ Wf,
                             int K, int N) {
  int wave = uwave();
  int lane = threadIdx.x & 31;
  int ntiles = N >> 4, ktiles = K >> 5;
  if (wave >= ntiles * ktiles) return;
  int kc = wave / ntiles, nt = wave % ntiles;
  int col = nt * 16 + (lane & 15);
  int kb  = kc * 32 + ((lane >= 16) ? 16 : 0);
  _Float16* dst = Wf + (size_t)wave * 512 + lane * 16;
#pragma unroll
  for (int i = 0; i < 16; ++i)
    dst[i] = (_Float16)W[(size_t)(kb + i) * N + col];
}

// ---------------------------------------------------------------------------
// GEMM: Out[M][N] (fp32, row-major) = X[M][K] (fp32) * Wf (f16 B-frag) + bias.
// One wave per 16x64 output strip (4 N-tiles): build the A-fragment once per
// K-chunk (two contiguous 8-float runs per lane -> contiguous k = 16*half + s, same order as Wf), load
// all 4 B fragments, then issue 4 back-to-back WMMAs into 4 accumulators.
// ---------------------------------------------------------------------------
__global__ void gemm_wmma_bias(const float* __restrict__ X, const _Float16* __restrict__ Wf,
                               const float* __restrict__ bias, float* __restrict__ Out,
                               int M, int N, int K) {
  int wave = uwave();
  int lane = threadIdx.x & 31;
  int ntiles  = N >> 4;
  int ngroups = N >> 6;           // groups of 4 n-tiles
  int mtiles  = M >> 4;
  if (wave >= mtiles * ngroups) return;
  int mt = wave / ngroups, ng = wave % ngroups;
  int lh = lane & 15;
  bool hi = lane >= 16;
  int laneoff = lane * 16;        // halfs into a fragment tile

  // divergent bases computed once; loop offsets are scalar/immediate
  const float* xrow = X + (size_t)(mt * 16) * K + (size_t)lh * K + (hi ? 16 : 0);
  int kchunks = K >> 5;
  v8f acc[4] = {};
  for (int kc = 0; kc < kchunks; ++kc) {
    const float* xp = xrow + kc * 32;
    v16h a;
#pragma unroll
    for (int i = 0; i < 8; ++i) a[i] = (_Float16)xp[i];
#pragma unroll
    for (int i = 0; i < 8; ++i) a[8 + i] = (_Float16)xp[8 + i];
    const _Float16* wk = Wf + (size_t)(kc * ntiles + ng * 4) * 512;  // uniform
    v16h b[4];
#pragma unroll
    for (int j = 0; j < 4; ++j)
      b[j] = *(const v16h*)(wk + j * 512 + laneoff);
#pragma unroll
    for (int j = 0; j < 4; ++j)
      acc[j] = WMMA_F16(a, b[j], acc[j]);
  }
  float* obase = Out + (size_t)(mt * 16 + (hi ? 8 : 0)) * N + ng * 64 + lh;
#pragma unroll
  for (int j = 0; j < 4; ++j) {
    float bv = bias ? bias[ng * 64 + j * 16 + lh] : 0.0f;
#pragma unroll
    for (int r = 0; r < 8; ++r)
      obase[(size_t)r * N + j * 16] = acc[j][r] + bv;
  }
}

// ---------------------------------------------------------------------------
// L2-normalize p0 per (head, row) over hd=32 and pack into f16 A-fragment
// layout per (n, head, qtile): lane l<16 -> row qt*16+l, K 0..15;
// lane l+16 -> same row, K 16..31.  Paired lanes share sumsq.
// ---------------------------------------------------------------------------
__global__ void pack_p0_afrag(const float* __restrict__ P0, _Float16* __restrict__ P0f,
                              int NB, int L) {
  int wave = uwave();
  int lane = threadIdx.x & 31;
  int qtiles = L >> 4;
  if (wave >= NB * 8 * qtiles) return;
  int qt = wave % qtiles;
  int nh = wave / qtiles;
  int h = nh % 8, nb = nh / 8;
  int row = qt * 16 + (lane & 15);
  const float* src = P0 + ((size_t)(nb * L + row)) * 256 + h * 32;
  int off = (lane >= 16) ? 16 : 0;
  float v[16];
  float ss = 0.0f;
#pragma unroll
  for (int i = 0; i < 8; ++i) { v[i] = src[off + i]; ss += v[i] * v[i]; }
#pragma unroll
  for (int i = 0; i < 8; ++i) { v[8 + i] = src[off + 8 + i]; ss += v[8 + i] * v[8 + i]; }
  ss += __shfl_xor(ss, 16, 32);
  float inv = 1.0f / fmaxf(sqrtf(ss), 1e-12f);
  _Float16* dst = P0f + (size_t)wave * 512 + lane * 16;
#pragma unroll
  for (int i = 0; i < 16; ++i) dst[i] = (_Float16)(v[i] * inv);
}

// ---------------------------------------------------------------------------
// From pv[M][512]: per head h, p1 = cols [h*64, h*64+32), v1 = [h*64+32, +64).
// Normalize p1 and pack into f16 B-fragment layout per (n, head, ktile):
// lane c<16 -> p1 row kt*16+c with hd 0..15; lane c+16 -> hd 16..31.
// Also copy v1 (fp32) to a contiguous (n,h,L,32) buffer.
// ---------------------------------------------------------------------------
__global__ void pack_pv(const float* __restrict__ PV, _Float16* __restrict__ P1f,
                        float* __restrict__ V1, int NB, int L) {
  int wave = uwave();
  int lane = threadIdx.x & 31;
  int ktiles = L >> 4;
  if (wave >= NB * 8 * ktiles) return;
  int kt = wave % ktiles;
  int nh = wave / ktiles;
  int h = nh % 8, nb = nh / 8;
  int r = kt * 16 + (lane & 15);
  const float* src = PV + ((size_t)(nb * L + r)) * 512 + h * 64;
  int off = (lane >= 16) ? 16 : 0;
  float v[16];
  float ss = 0.0f;
#pragma unroll
  for (int i = 0; i < 16; ++i) { v[i] = src[off + i]; ss += v[i] * v[i]; }
  ss += __shfl_xor(ss, 16, 32);
  float inv = 1.0f / fmaxf(sqrtf(ss), 1e-12f);
  _Float16* dst = P1f + (size_t)wave * 512 + lane * 16;
#pragma unroll
  for (int i = 0; i < 16; ++i) dst[i] = (_Float16)(v[i] * inv);
  // v1 copy (fp32)
  float* vdst = V1 + ((size_t)nh * L + r) * 32 + off;
#pragma unroll
  for (int i = 0; i < 16; ++i) vdst[i] = src[32 + off + i];
}

// ---------------------------------------------------------------------------
// Fused sim + max/argmax: one wave per (n, head, 4 q-tiles).
// 4 A-fragments resident; loop over 144 k-tiles: ONE v16h B load (scalar
// walking pointer + per-lane 32B offset) feeds FOUR WMMAs (16x16x32, K = hd
// exactly).  Running per-lane max/argmax of alpha*s+beta (mask applied), then
// 16-lane butterfly reduce (max, first-index tie-break).  Sigmoid applied
// once at the end (monotonic -> commutes with max/argmax); the 680 MB sim
// matrix is never materialized.
// ---------------------------------------------------------------------------
__global__ void sim_max_argmax(const _Float16* __restrict__ P0f, const _Float16* __restrict__ P1f,
                               const unsigned char* __restrict__ Mask,
                               const float* __restrict__ alpha, const float* __restrict__ beta,
                               float* __restrict__ MaxSim, int* __restrict__ ArgMax,
                               int NB, int L) {
  int wave = uwave();
  int lane = threadIdx.x & 31;
  int qtiles = L >> 4;
  int qgroups = qtiles >> 2;          // 4 q-tiles per wave
  if (wave >= NB * 8 * qgroups) return;
  int qg = wave % qgroups;
  int nh = wave / qgroups;
  int nb = nh / 8;
  float al = alpha[0], be = beta[0];
  int laneoff = lane * 16;            // halfs into a fragment tile

  const _Float16* p0base = P0f + (size_t)nh * qtiles * 512;   // 512 halfs / tile
  const _Float16* bt     = P1f + (size_t)nh * qtiles * 512;   // scalar walker
  const unsigned char* mp = Mask + (size_t)nb * L;            // scalar walker
  int col = lane & 15;

  v16h a[4];
#pragma unroll
  for (int j = 0; j < 4; ++j)
    a[j] = *(const v16h*)(p0base + (size_t)(qg * 4 + j) * 512 + laneoff);

  float bmax[4][8];
  int   barg[4][8];
#pragma unroll
  for (int j = 0; j < 4; ++j)
#pragma unroll
    for (int i = 0; i < 8; ++i) { bmax[j][i] = -INFINITY; barg[j][i] = 0; }

  for (int kt = 0; kt < qtiles; ++kt) {
    if (kt + 1 < qtiles) __builtin_prefetch(bt + 512 + laneoff, 0, 1);
    v16h b = *(const v16h*)(bt + laneoff);
    bool mk = mp[col] != 0;
    int k = kt * 16 + col;
#pragma unroll
    for (int j = 0; j < 4; ++j) {
      v8f c = {};
      c = WMMA_F16(a[j], b, c);
#pragma unroll
      for (int i = 0; i < 8; ++i) {
        float s = mk ? (al * c[i] + be) : -INFINITY;
        if (s > bmax[j][i]) { bmax[j][i] = s; barg[j][i] = k; }  // strict > keeps first
      }
    }
    bt += 512;
    mp += 16;
  }
  // reduce across the 16 columns (xor 8,4,2,1 stays within each 16-lane half)
#pragma unroll
  for (int d = 8; d >= 1; d >>= 1) {
#pragma unroll
    for (int j = 0; j < 4; ++j)
#pragma unroll
      for (int i = 0; i < 8; ++i) {
        float om = __shfl_xor(bmax[j][i], d, 32);
        int   oa = __shfl_xor(barg[j][i], d, 32);
        if (om > bmax[j][i] || (om == bmax[j][i] && oa < barg[j][i])) {
          bmax[j][i] = om; barg[j][i] = oa;
        }
      }
  }
  if ((lane & 15) == 0) {
    float* mo = MaxSim + (size_t)nh * L;
    int*   ao = ArgMax + (size_t)nh * L;
#pragma unroll
    for (int j = 0; j < 4; ++j) {
      int mbase = (qg * 4 + j) * 16 + ((lane >= 16) ? 8 : 0);
#pragma unroll
      for (int i = 0; i < 8; ++i) {
        mo[mbase + i] = 1.0f / (1.0f + __expf(-bmax[j][i]));  // sigmoid(max)==max(sigmoid)
        ao[mbase + i] = barg[j][i];
      }
    }
  }
}

// ---------------------------------------------------------------------------
// message[n, q, h*32+j] = max_sim[n,h,q] * v1[n,h,argmax[n,h,q], j]
// ---------------------------------------------------------------------------
__global__ void build_message(const float* __restrict__ V1, const float* __restrict__ MaxSim,
                              const int* __restrict__ ArgMax, float* __restrict__ Msg,
                              int NB, int L) {
  int tid = blockIdx.x * blockDim.x + threadIdx.x;
  int total = NB * 8 * L * 32;
  if (tid >= total) return;
  int j = tid & 31;
  int rest = tid >> 5;
  int q = rest % L;
  int nh = rest / L;
  int h = nh & 7, nb = nh >> 3;
  float ms = MaxSim[(size_t)nh * L + q];
  int idx = ArgMax[(size_t)nh * L + q];
  float val = ms * V1[((size_t)nh * L + idx) * 32 + j];
  Msg[((size_t)(nb * L + q)) * 256 + h * 32 + j] = val;
}

// ---------------------------------------------------------------------------
extern "C" void kernel_launch(void* const* d_in, const int* in_sizes, int n_in,
                              void* d_out, int out_size, void* d_ws, size_t ws_size,
                              hipStream_t stream) {
  constexpr int NB = 4, L = 2304, DIM = 256;
  constexpr int M = NB * L;           // 9216 rows
  constexpr int QT = L / 16;          // 144 tiles

  const float*         x0    = (const float*)d_in[0];
  const float*         x1    = (const float*)d_in[1];
  const unsigned char* mask  = (const unsigned char*)d_in[2];
  const float*         W0    = (const float*)d_in[3];
  const float*         b0    = (const float*)d_in[4];
  const float*         W1    = (const float*)d_in[5];
  const float*         b1    = (const float*)d_in[6];
  const float*         Wo    = (const float*)d_in[7];
  const float*         bo    = (const float*)d_in[8];
  const float*         alpha = (const float*)d_in[9];
  const float*         beta  = (const float*)d_in[10];
  float*               out   = (float*)d_out;

  char* ws = (char*)d_ws;
  size_t off = 0;
  auto alloc = [&](size_t bytes) -> char* {
    char* p = ws + off;
    off = (off + bytes + 255) & ~(size_t)255;
    return p;
  };
  float*     p0     = (float*)alloc((size_t)M * DIM * sizeof(float));
  float*     pv     = (float*)alloc((size_t)M * 2 * DIM * sizeof(float));
  _Float16*  p0f    = (_Float16*)alloc((size_t)NB * 8 * QT * 512 * sizeof(_Float16));
  _Float16*  p1f    = (_Float16*)alloc((size_t)NB * 8 * QT * 512 * sizeof(_Float16));
  float*     v1     = (float*)alloc((size_t)NB * 8 * L * 32 * sizeof(float));
  float*     maxsim = (float*)alloc((size_t)NB * 8 * L * sizeof(float));
  int*       argmax = (int*)alloc((size_t)NB * 8 * L * sizeof(int));
  float*     msg    = (float*)alloc((size_t)M * DIM * sizeof(float));
  _Float16*  w0f    = (_Float16*)alloc((size_t)DIM * DIM * sizeof(_Float16));
  _Float16*  w1f    = (_Float16*)alloc((size_t)DIM * 2 * DIM * sizeof(_Float16));
  _Float16*  wof    = (_Float16*)alloc((size_t)DIM * DIM * sizeof(_Float16));

  const int TPB = 256;  // 8 waves/block
  auto blocks_for_waves = [](int waves) { return (waves * 32 + 255) / 256; };

  // 1) Weight pre-swizzle into B-fragment layouts (one-time, tiny)
  pack_w_bfrag<<<blocks_for_waves((DIM / 32) * (DIM / 16)), TPB, 0, stream>>>(W0, w0f, DIM, DIM);
  pack_w_bfrag<<<blocks_for_waves((DIM / 32) * (2 * DIM / 16)), TPB, 0, stream>>>(W1, w1f, DIM, 2 * DIM);
  pack_w_bfrag<<<blocks_for_waves((DIM / 32) * (DIM / 16)), TPB, 0, stream>>>(Wo, wof, DIM, DIM);

  // 2) Projection GEMMs (WMMA, 4 N-tiles per wave)
  gemm_wmma_bias<<<blocks_for_waves((M / 16) * (DIM / 64)), TPB, 0, stream>>>(
      x0, w0f, b0, p0, M, DIM, DIM);
  gemm_wmma_bias<<<blocks_for_waves((M / 16) * (2 * DIM / 64)), TPB, 0, stream>>>(
      x1, w1f, b1, pv, M, 2 * DIM, DIM);

  // 3) Normalize + pack into WMMA fragment layouts
  pack_p0_afrag<<<blocks_for_waves(NB * 8 * QT), TPB, 0, stream>>>(p0, p0f, NB, L);
  pack_pv<<<blocks_for_waves(NB * 8 * QT), TPB, 0, stream>>>(pv, p1f, v1, NB, L);

  // 4) Fused sim GEMM + masked max/argmax + sigmoid (4 q-tiles per wave)
  sim_max_argmax<<<blocks_for_waves(NB * 8 * (QT / 4)), TPB, 0, stream>>>(
      p0f, p1f, mask, alpha, beta, maxsim, argmax, NB, L);

  // 5) Gather + scale into message
  build_message<<<(NB * 8 * L * 32 + TPB - 1) / TPB, TPB, 0, stream>>>(
      v1, maxsim, argmax, msg, NB, L);

  // 6) Output GEMM (WMMA) directly into d_out
  gemm_wmma_bias<<<blocks_for_waves((M / 16) * (DIM / 64)), TPB, 0, stream>>>(
      msg, wof, bo, out, M, DIM, DIM);
}
